// SortPool_82901458747987
// MI455X (gfx1250) — matrix operand order repaired and checked by reading, not compiled, not run on hardware
//
#include <hip/hip_runtime.h>
#include <hip/hip_bf16.h>

#define N_NODES 131072
#define N_PER   1024
#define B_GR    128
#define N_EDGES 2097152
#define K_POOL  30
#define HID     128
#define IN_CH   14
#define OUT_DIM 4

typedef __attribute__((ext_vector_type(16))) _Float16 v16h;
typedef __attribute__((ext_vector_type(8)))  float    v8f;

// ---------------------------------------------------------------- utilities
__global__ void sp_zero_f32(float* __restrict__ p, long long n) {
  long long i = (long long)blockIdx.x * blockDim.x + threadIdx.x;
  if (i < n) p[i] = 0.0f;
}

__global__ void sp_deg_count(const int* __restrict__ dst, float* __restrict__ deg) {
  int e = blockIdx.x * blockDim.x + threadIdx.x;
  if (e < N_EDGES) atomicAdd(&deg[dst[e]], 1.0f);
}

__global__ void sp_deg_inv(float* __restrict__ deg) {
  int i = blockIdx.x * blockDim.x + threadIdx.x;
  if (i < N_NODES) {
    float d = deg[i];
    deg[i] = (d > 0.0f) ? (1.0f / d) : 0.0f;
  }
}

// -------------------------------------------------------- edge aggregation
// generic (layer 1, C=14): one thread per (edge, channel)
__global__ void sp_aggregate(const float* __restrict__ xin,
                             const int* __restrict__ src,
                             const int* __restrict__ dst,
                             float* __restrict__ agg,
                             int C, long long total) {
  long long idx = (long long)blockIdx.x * blockDim.x + threadIdx.x;
  if (idx >= total) return;
  int e = (int)(idx / C), c = (int)(idx % C);
  atomicAdd(&agg[(size_t)dst[e] * C + c], xin[(size_t)src[e] * C + c]);
}

// C=128: one thread per (edge, 4 channels) -> b128 gather + 4 atomics
__global__ void sp_aggregate4(const float* __restrict__ xin,
                              const int* __restrict__ src,
                              const int* __restrict__ dst,
                              float* __restrict__ agg) {
  long long idx = (long long)blockIdx.x * blockDim.x + threadIdx.x;
  if (idx >= (long long)N_EDGES * 32) return;
  int e = (int)(idx >> 5);
  int q = (int)((idx & 31) << 2);
  const float4 v = *(const float4*)(xin + (size_t)src[e] * HID + q);
  float* a = agg + (size_t)dst[e] * HID + q;
  atomicAdd(a + 0, v.x);
  atomicAdd(a + 1, v.y);
  atomicAdd(a + 2, v.z);
  atomicAdd(a + 3, v.w);
}

// --------------------------------------------- pack B = [Wl;Wr]^T fragments
// Layout: Bp[((ntile*nk + kb)*32 + lane)*16 + h]  with
//   n = ntile*16 + lane%16 ,  k = kb*32 + 8*(lane/16) + (h%8) + 16*(h/8)
// (16-bit B 32x16 WMMA lane layout -> each lane loads 32 contiguous bytes)
__global__ void sp_pack_w(const float* __restrict__ wl,
                          const float* __restrict__ wr,
                          _Float16* __restrict__ Bp, int C, int Kpad) {
  int nk = Kpad >> 5;
  int total = 8 * nk * 32 * 16;          // = Kpad * 128
  int idx = blockIdx.x * blockDim.x + threadIdx.x;
  if (idx >= total) return;
  int h     = idx & 15;
  int lane  = (idx >> 4) & 31;
  int kb    = (idx >> 9) % nk;
  int ntile = idx / (nk << 9);
  int n = ntile * 16 + (lane & 15);
  int k = kb * 32 + ((lane >> 4) << 3) + (h & 7) + ((h >> 3) << 4);
  float v = 0.0f;
  if (k < C)          v = wl[n * C + k];
  else if (k < 2 * C) v = wr[n * C + (k - C)];
  Bp[idx] = (_Float16)v;
}

// --------------------------------------------------------------- WMMA GEMM
// A-fragment builder, 16-bit A 16x32 layout:
//   lane group g: halves 0..7 -> k = kb*32+8g .. +7 ; halves 8..15 -> +16
template<int C>
__device__ __forceinline__ v16h sp_makeA(const float* __restrict__ aggRow,
                                         const float* __restrict__ xRow,
                                         float s, int kb, int g) {
  union { v16h v; _Float16 h[16]; } f;
  if constexpr (C == 128) {
    // chunks of 8 are aligned and never straddle the agg|x boundary
    const int k0 = kb * 32 + 8 * g;
    const int k1 = k0 + 16;
    const float* p0; float s0;
    if (k0 < C) { p0 = aggRow + k0; s0 = s; } else { p0 = xRow + (k0 - C); s0 = 1.0f; }
    const float* p1; float s1;
    if (k1 < C) { p1 = aggRow + k1; s1 = s; } else { p1 = xRow + (k1 - C); s1 = 1.0f; }
    float4 a = *(const float4*)p0, b = *(const float4*)(p0 + 4);
    f.h[0]=(_Float16)(a.x*s0); f.h[1]=(_Float16)(a.y*s0);
    f.h[2]=(_Float16)(a.z*s0); f.h[3]=(_Float16)(a.w*s0);
    f.h[4]=(_Float16)(b.x*s0); f.h[5]=(_Float16)(b.y*s0);
    f.h[6]=(_Float16)(b.z*s0); f.h[7]=(_Float16)(b.w*s0);
    a = *(const float4*)p1; b = *(const float4*)(p1 + 4);
    f.h[8]=(_Float16)(a.x*s1);  f.h[9]=(_Float16)(a.y*s1);
    f.h[10]=(_Float16)(a.z*s1); f.h[11]=(_Float16)(a.w*s1);
    f.h[12]=(_Float16)(b.x*s1); f.h[13]=(_Float16)(b.y*s1);
    f.h[14]=(_Float16)(b.z*s1); f.h[15]=(_Float16)(b.w*s1);
  } else {
    // layer 1 (C=14, Kpad=32, kb==0): branch-free, always-in-bounds loads.
    // halves 0..7: k in [8g, 8g+7]  -> agg (k<C) or x (k>=C, only g=1,k=14,15)
    #pragma unroll
    for (int i = 0; i < 8; ++i) {
      int k = 8 * g + i;                       // <= 15; agg row over-read is safe
      float va = aggRow[k] * s;
      float vx = xRow[k >= C ? k - C : 0];
      f.h[i] = (_Float16)((k < C) ? va : vx);
    }
    // halves 8..15: k2 in [16+8g, 23+8g] -> x (k2<2C) or zero pad
    #pragma unroll
    for (int i = 0; i < 8; ++i) {
      int k2 = 16 + 8 * g + i;
      int xi = k2 - C;                         // in [2,17]
      float vv = xRow[xi < C ? xi : C - 1];    // clamped, always in-bounds
      f.h[8 + i] = (_Float16)((k2 < 2 * C) ? vv : 0.0f);
    }
  }
  return f.v;
}

// out[N,128] = relu( [agg*dinv | x] @ Bp + bias )
// one wave -> 32x16 output tile (two row sub-tiles share each B fragment)
template<int C, int Kpad>
__global__ __launch_bounds__(256)
void sp_gemm_wmma(const float* __restrict__ agg, const float* __restrict__ xin,
                  const float* __restrict__ dinv,
                  const _Float16* __restrict__ Bp,
                  const float* __restrict__ bias, float* __restrict__ out) {
  constexpr int NK = Kpad / 32;
  const int lane  = threadIdx.x;             // 0..31
  const int wtile = threadIdx.y;             // column tile 0..7
  const int g  = (lane >> 4) & 1;            // bounded so k-compares const-fold
  const int mr = lane & 15;
  const int rbase = blockIdx.x * 32;
  const int row0 = rbase + mr;
  const int row1 = row0 + 16;
  const float sc0 = dinv[row0], sc1 = dinv[row1];
  const float* aggR0 = agg + (size_t)row0 * C;
  const float* xR0   = xin + (size_t)row0 * C;
  const float* aggR1 = agg + (size_t)row1 * C;
  const float* xR1   = xin + (size_t)row1 * C;

  v8f acc0 = {}, acc1 = {};
  #pragma unroll
  for (int kb = 0; kb < NK; ++kb) {
    const _Float16* bptr = Bp + ((((size_t)wtile * NK + kb) * 32 + lane) * 16);
    v16h bf = *(const v16h*)bptr;
    v16h a0 = sp_makeA<C>(aggR0, xR0, sc0, kb, g);
    v16h a1 = sp_makeA<C>(aggR1, xR1, sc1, kb, g);
    acc0 = __builtin_amdgcn_wmma_f32_16x16x32_f16(
        false, a0, false, bf, (short)0, acc0, false, false);
    acc1 = __builtin_amdgcn_wmma_f32_16x16x32_f16(
        false, a1, false, bf, (short)0, acc1, false, false);
  }

  // C/D layout: VGPR r -> M = r + 8*(lane/16), N = lane%16
  const int col = wtile * 16 + mr;
  const float bv = bias[col];
  #pragma unroll
  for (int r = 0; r < 8; ++r) {
    float v0 = acc0[r] + bv;
    out[(size_t)(rbase + r + 8 * g) * HID + col] = (v0 > 0.0f) ? v0 : 0.0f;
    float v1 = acc1[r] + bv;
    out[(size_t)(rbase + 16 + r + 8 * g) * HID + col] = (v1 > 0.0f) ? v1 : 0.0f;
  }
}

// --------------------------------------------------------------- sort-pool
// Per graph: rank nodes by last channel (desc, index-asc tiebreak), gather top-30.
__global__ __launch_bounds__(1024)
void sp_sortpool(const float* __restrict__ h, float* __restrict__ top) {
  __shared__ float vals[N_PER];
  __shared__ int   sel[K_POOL];
  const int b = blockIdx.x, t = threadIdx.x;
  const int node = b * N_PER + t;
  const float v = h[(size_t)node * HID + (HID - 1)];
  vals[t] = v;
  __syncthreads();
  int rank = 0;
  for (int j = 0; j < N_PER; ++j) {
    float vj = vals[j];
    rank += (vj > v) || (vj == v && j < t);
  }
  if (rank < K_POOL) sel[rank] = node;
  __syncthreads();
  for (int i = t; i < K_POOL * HID; i += N_PER) {
    int s = i >> 7, c = i & (HID - 1);
    top[((size_t)b * K_POOL + s) * HID + c] = h[(size_t)sel[s] * HID + c];
  }
}

// --------------------------------------------------- conv1d + lin1 + lin2
__global__ __launch_bounds__(256)
void sp_head(const float* __restrict__ top, const float* __restrict__ cw,
             const float* __restrict__ cb, const float* __restrict__ l1w,
             const float* __restrict__ l1b, const float* __restrict__ l2w,
             const float* __restrict__ l2b, float* __restrict__ out) {
  __shared__ float tl[K_POOL * HID];   // 30x128
  __shared__ float yf[32 * 26];        // conv output, flatten order ch*26+t
  __shared__ float z[HID];
  const int b = blockIdx.x, t = threadIdx.x;
  for (int i = t; i < K_POOL * HID; i += 256)
    tl[i] = top[(size_t)b * K_POOL * HID + i];
  __syncthreads();
  for (int idx = t; idx < 32 * 26; idx += 256) {
    int ch = idx / 26, tt = idx % 26;
    float s = cb[ch];
    for (int kk = 0; kk < 5; ++kk) {
      const float* wrow = cw + ch * (HID * 5) + kk;   // w[ch][c][kk]
      const float* trow = tl + (tt + kk) * HID;
      #pragma unroll 4
      for (int c = 0; c < HID; ++c) s += trow[c] * wrow[c * 5];
    }
    yf[ch * 26 + tt] = (s > 0.0f) ? s : 0.0f;
  }
  __syncthreads();
  if (t < HID) {
    float s = l1b[t];
    const float* w = l1w + t * (32 * 26);
    #pragma unroll 4
    for (int i = 0; i < 32 * 26; ++i) s += yf[i] * w[i];
    z[t] = (s > 0.0f) ? s : 0.0f;
  }
  __syncthreads();
  if (t < OUT_DIM) {
    float s = l2b[t];
    const float* w = l2w + t * HID;
    #pragma unroll 4
    for (int c = 0; c < HID; ++c) s += z[c] * w[c];
    out[b * OUT_DIM + t] = s;
  }
}

// ------------------------------------------------------------------- launch
extern "C" void kernel_launch(void* const* d_in, const int* in_sizes, int n_in,
                              void* d_out, int out_size, void* d_ws, size_t ws_size,
                              hipStream_t stream) {
  const float* x    = (const float*)d_in[0];
  const int*   ei   = (const int*)d_in[1];
  const int*   src  = ei;
  const int*   dst  = ei + N_EDGES;
  const float* w_l1 = (const float*)d_in[3];
  const float* b_l1 = (const float*)d_in[4];
  const float* w_r1 = (const float*)d_in[5];
  const float* w_l2 = (const float*)d_in[6];
  const float* b_l2 = (const float*)d_in[7];
  const float* w_r2 = (const float*)d_in[8];
  const float* w_l3 = (const float*)d_in[9];
  const float* b_l3 = (const float*)d_in[10];
  const float* w_r3 = (const float*)d_in[11];
  const float* cw   = (const float*)d_in[12];
  const float* cb   = (const float*)d_in[13];
  const float* l1w  = (const float*)d_in[14];
  const float* l1b  = (const float*)d_in[15];
  const float* l2w  = (const float*)d_in[16];
  const float* l2b  = (const float*)d_in[17];
  float* out = (float*)d_out;

  // workspace carve-up (all naturally aligned, d_ws is 256B-aligned)
  float*    deg = (float*)d_ws;                            // N (becomes deg_inv)
  float*    agg = deg + N_NODES;                           // N*128
  float*    hA  = agg + (size_t)N_NODES * HID;             // N*128
  float*    hB  = hA  + (size_t)N_NODES * HID;             // N*128
  _Float16* Bp  = (_Float16*)(hB + (size_t)N_NODES * HID); // 256*128 halves
  float*    top = (float*)(Bp + 256 * HID);                // B*30*128

  const int TB = 256;
  auto blocks = [](long long n, int tb) { return (unsigned)((n + tb - 1) / tb); };

  // degrees -> deg_inv (in place)
  sp_zero_f32<<<blocks(N_NODES, TB), TB, 0, stream>>>(deg, N_NODES);
  sp_deg_count<<<blocks(N_EDGES, TB), TB, 0, stream>>>(dst, deg);
  sp_deg_inv<<<blocks(N_NODES, TB), TB, 0, stream>>>(deg);

  // ---- layer 1: C=14, Kpad=32
  {
    long long tot = (long long)N_EDGES * IN_CH;
    sp_zero_f32<<<blocks((long long)N_NODES * IN_CH, TB), TB, 0, stream>>>(
        agg, (long long)N_NODES * IN_CH);
    sp_aggregate<<<blocks(tot, TB), TB, 0, stream>>>(x, src, dst, agg, IN_CH, tot);
    sp_pack_w<<<blocks(32 * HID, TB), TB, 0, stream>>>(w_l1, w_r1, Bp, IN_CH, 32);
    sp_gemm_wmma<IN_CH, 32><<<N_NODES / 32, dim3(32, 8), 0, stream>>>(
        agg, x, deg, Bp, b_l1, hA);
  }
  // ---- layer 2: C=128, Kpad=256
  {
    long long tot = (long long)N_EDGES * 32;
    sp_zero_f32<<<blocks((long long)N_NODES * HID, TB), TB, 0, stream>>>(
        agg, (long long)N_NODES * HID);
    sp_aggregate4<<<blocks(tot, TB), TB, 0, stream>>>(hA, src, dst, agg);
    sp_pack_w<<<blocks(256 * HID, TB), TB, 0, stream>>>(w_l2, w_r2, Bp, HID, 256);
    sp_gemm_wmma<HID, 256><<<N_NODES / 32, dim3(32, 8), 0, stream>>>(
        agg, hA, deg, Bp, b_l2, hB);
  }
  // ---- layer 3: C=128, Kpad=256 (hB -> hA)
  {
    long long tot = (long long)N_EDGES * 32;
    sp_zero_f32<<<blocks((long long)N_NODES * HID, TB), TB, 0, stream>>>(
        agg, (long long)N_NODES * HID);
    sp_aggregate4<<<blocks(tot, TB), TB, 0, stream>>>(hB, src, dst, agg);
    sp_pack_w<<<blocks(256 * HID, TB), TB, 0, stream>>>(w_l3, w_r3, Bp, HID, 256);
    sp_gemm_wmma<HID, 256><<<N_NODES / 32, dim3(32, 8), 0, stream>>>(
        agg, hB, deg, Bp, b_l3, hA);
  }

  // sort-pool + head
  sp_sortpool<<<B_GR, N_PER, 0, stream>>>(hA, top);
  sp_head<<<B_GR, 256, 0, stream>>>(top, cw, cb, l1w, l1b, l2w, l2b, out);
}
